// BaseRNN_44985487459142
// MI455X (gfx1250) — matrix operand order repaired and checked by reading, not compiled, not run on hardware
//
#include <hip/hip_runtime.h>

// Vanilla tanh-RNN scan for MI455X (gfx1250, wave32, WMMA).
// seq=2048, batch=64, input=hidden=128.
//
// Batch rows are independent through the recurrence -> 4 workgroups, each
// owning a 16-row batch slice, run private serial scans (no grid sync).
// Per WG: 8 waves; wave w computes the 16x16 tile (slice rows 0..15, hidden
// cols 16w..16w+15). Weights are loop-invariant f16 B-fragments in registers.
//
// Software pipeline (per iteration t):
//   - stage x_{t+2} (f32->f16) into the ping-pong LDS x buffer
//   - CRITICAL CHAIN: acc = accx(t);  4x v_wmma (h_t @ Wh);  v_tanh;  write h
//   - INDEPENDENT:    accx(t+1) = bias + x_{t+1} @ Wx  (separate acc chain,
//                     overlaps with the critical chain in the matrix pipe)
//   - one __syncthreads(); ping-pong buffers avoid WAR hazards.

#define SEQ   2048
#define BATCH 64
#define NIN   128
#define NH    128

typedef __attribute__((ext_vector_type(16))) _Float16 v16h;
typedef __attribute__((ext_vector_type(8)))  float    v8f;

// Hardware tanh (V_TANH_F32) when available; branch-free fallback otherwise.
static __device__ __forceinline__ float fast_tanhf(float x) {
#if defined(__AMDGCN__) && __has_builtin(__builtin_amdgcn_tanhf)
  return __builtin_amdgcn_tanhf(x);
#elif defined(__AMDGCN__) && __has_builtin(__builtin_amdgcn_tanh_f32)
  return __builtin_amdgcn_tanh_f32(x);
#elif defined(__AMDGCN__)
  // tanh(x) = 1 - 2/(e^{2x}+1); clamp keeps exp finite, path is branch-free
  // (v_exp_f32 + v_rcp_f32, no saveexec divergence inside the scan loop).
  const float xc = fminf(fmaxf(x, -9.0f), 9.0f);
  const float e  = __expf(2.0f * xc);
  return 1.0f - 2.0f * __builtin_amdgcn_rcpf(e + 1.0f);
#else
  return tanhf(x);  // host pass only; never executed on device
#endif
}

// 16x32 f16 A-fragment from a row-major f16 LDS tile [16][128].
// Lane L: row m=L&15; lanes 0-15 hold K={0..7,16..23}, lanes 16-31 hold
// K={8..15,24..31}; each VGPR is a consecutive K-pair -> b32 reads that the
// compiler merges into ds_load_b128.
static __device__ __forceinline__ v16h load_a_frag(const _Float16* buf, int lane, int kblk) {
  const int m  = lane & 15;
  const int kb = kblk * 32 + ((lane >> 4) << 3);
  const unsigned int* b32 = (const unsigned int*)(buf + m * 128);
  union { unsigned int u[8]; v16h h; } r;
#pragma unroll
  for (int j = 0; j < 4; ++j) {
    r.u[j]     = b32[(kb + 2 * j) >> 1];
    r.u[j + 4] = b32[(kb + 16 + 2 * j) >> 1];
  }
  return r.h;
}

// 32x16 f16 B-fragment from row-major f32 weights W[k][n] (128x128).
// Lane L: column n = nbase + (L&15); lanes 0-15 carry K={0..7,16..23}, lanes 16-31
// carry K={8..15,24..31} of the 32-deep k-block. One-time setup cost.
static __device__ __forceinline__ v16h load_b_frag(const float* W, int lane, int kblk, int nbase) {
  const int n  = nbase + (lane & 15);
  const int k0 = kblk * 32 + ((lane >> 4) << 3);
  v16h r;
#pragma unroll
  for (int i = 0; i < 16; ++i)
    r[i] = (_Float16)W[(k0 + i + ((i >> 3) << 3)) * NH + n];
  return r;
}

// Stage a 16x128 f32 tile from global into LDS as f16 (256 thr x 8 elems).
static __device__ __forceinline__ void stage_tile_f16(_Float16* dst, const float* src, int tid) {
  const float4* s4 = (const float4*)src;
  float4 a = s4[2 * tid];
  float4 b = s4[2 * tid + 1];
  union { _Float16 h[8]; uint4 v; } p;
  p.h[0] = (_Float16)a.x; p.h[1] = (_Float16)a.y;
  p.h[2] = (_Float16)a.z; p.h[3] = (_Float16)a.w;
  p.h[4] = (_Float16)b.x; p.h[5] = (_Float16)b.y;
  p.h[6] = (_Float16)b.z; p.h[7] = (_Float16)b.w;
  *(uint4*)(dst + 8 * tid) = p.v;
}

__global__ __launch_bounds__(256, 1)
void rnn_scan_wmma(const float* __restrict__ X,     // (SEQ, BATCH, NIN)
                   const float* __restrict__ H0,    // (BATCH, NH)
                   const float* __restrict__ Wx,    // (NIN, NH)
                   const float* __restrict__ Wh,    // (NH, NH)
                   const float* __restrict__ bias,  // (NH)
                   float* __restrict__ out)         // outputs ++ initial hidden
{
  __shared__ __align__(16) _Float16 xbuf[2][16 * NIN];  // ping-pong, depth-2 staged
  __shared__ __align__(16) _Float16 hbuf[2][16 * NH];   // ping-pong recurrent state

  const int tid   = threadIdx.x;
  const int lane  = tid & 31;
  const int wave  = tid >> 5;          // 0..7 -> hidden-column tile
  const int nbase = wave * 16;
  const int row0  = blockIdx.x * 16;   // batch-row slice base (4 blocks)

  // Loop-invariant weight fragments in registers (f32 -> f16 once).
  v16h Bx[4], Bh[4];
#pragma unroll
  for (int kb = 0; kb < 4; ++kb) {
    Bx[kb] = load_b_frag(Wx, lane, kb, nbase);
    Bh[kb] = load_b_frag(Wh, lane, kb, nbase);
  }

  const float bval = bias[nbase + (lane & 15)];
  const v8f bias8 = { bval, bval, bval, bval, bval, bval, bval, bval };

  // Second reference output: the INITIAL hidden state (source bug), appended
  // after the (SEQ,BATCH,NH) outputs. Each WG copies its 16-row slice.
  {
    const float* src = H0 + (size_t)row0 * NH;
    float* dst = out + (size_t)SEQ * BATCH * NH + (size_t)row0 * NH;
    for (int i = tid; i < 16 * NH; i += 256) dst[i] = src[i];
  }

  // Prologue: stage x0, x1, h0; precompute accx(0) = bias + x0 @ Wx.
  stage_tile_f16(xbuf[0], X + (size_t)row0 * NIN, tid);                     // x(0)
  stage_tile_f16(xbuf[1], X + ((size_t)BATCH + row0) * NIN, tid);           // x(1)
  stage_tile_f16(hbuf[0], H0 + (size_t)row0 * NH, tid);
  __syncthreads();

  v8f accx = bias8;
#pragma unroll
  for (int kb = 0; kb < 4; ++kb) {
    v16h a = load_a_frag(xbuf[0], lane, kb);
    accx = __builtin_amdgcn_wmma_f32_16x16x32_f16(false, a, false, Bx[kb],
                                                  (short)0, accx, false, false);
  }
  __syncthreads();  // xbuf[0] is overwritten at t=0; all reads must be done

  for (int t = 0; t < SEQ; ++t) {
    const int cur = t & 1, nxt = cur ^ 1;

    // Stage x_{t+2} into the buffer whose x_t contents are now dead.
    if (t + 2 < SEQ)
      stage_tile_f16(xbuf[cur], X + ((size_t)(t + 2) * BATCH + row0) * NIN, tid);
    if (t + 4 < SEQ)
      __builtin_prefetch(X + ((size_t)(t + 4) * BATCH + row0) * NIN + tid * 8, 0, 1);

    // CRITICAL CHAIN: acc = accx(t) + h_t @ Wh.
    v8f acc = accx;
#pragma unroll
    for (int kb = 0; kb < 4; ++kb) {
      v16h a = load_a_frag(hbuf[cur], lane, kb);
      acc = __builtin_amdgcn_wmma_f32_16x16x32_f16(false, a, false, Bh[kb],
                                                   (short)0, acc, false, false);
    }

    // INDEPENDENT CHAIN: accx(t+1) = bias + x_{t+1} @ Wx (overlaps h-chain).
    v8f accx_n = bias8;
    if (t + 1 < SEQ) {
#pragma unroll
      for (int kb = 0; kb < 4; ++kb) {
        v16h a = load_a_frag(xbuf[nxt], lane, kb);
        accx_n = __builtin_amdgcn_wmma_f32_16x16x32_f16(false, a, false, Bx[kb],
                                                        (short)0, accx_n, false, false);
      }
    }

    // v_tanh_f32; write f32 output tile and f16 next-h tile.
    // C/D layout: VGPR r, lane L -> (m = r + 8*(L>>4), n = nbase + (L&15)).
    const int nc  = nbase + (lane & 15);
    const int mhi = (lane >> 4) << 3;
    float* orow = out + ((size_t)t * BATCH + row0) * NH;
#pragma unroll
    for (int r = 0; r < 8; ++r) {
      const int m = r + mhi;
      const float v = fast_tanhf(acc[r]);
      orow[(size_t)m * NH + nc] = v;
      hbuf[nxt][m * NH + nc] = (_Float16)v;
    }
    __syncthreads();   // single barrier per step
    accx = accx_n;
  }
}

extern "C" void kernel_launch(void* const* d_in, const int* in_sizes, int n_in,
                              void* d_out, int out_size, void* d_ws, size_t ws_size,
                              hipStream_t stream) {
  const float* X  = (const float*)d_in[0];   // inputs (2048,64,128)
  const float* H0 = (const float*)d_in[1];   // hidden (64,128)
  const float* Wx = (const float*)d_in[2];   // (128,128)
  const float* Wh = (const float*)d_in[3];   // (128,128)
  const float* b  = (const float*)d_in[4];   // (128)
  float* out = (float*)d_out;
  rnn_scan_wmma<<<dim3(BATCH / 16), dim3(256), 0, stream>>>(X, H0, Wx, Wh, b, out);
}
